// SimpleDAGNN_46694884442364
// MI455X (gfx1250) — compile-verified, matrix operand order checked by reading, not run on hardware
//
#include <hip/hip_runtime.h>
#include <hip/hip_bf16.h>
#include <math.h>

// ---------------------------------------------------------------------------
// CDNA5 (gfx1250) wave32 WMMA implementation of the DAG-GNN reference.
// GEMMs use v_wmma_f32_16x16x32_f16 (f16 inputs, f32 accumulate).
// ---------------------------------------------------------------------------

typedef __attribute__((ext_vector_type(16))) _Float16 v16h;
typedef __attribute__((ext_vector_type(8)))  float    v8f;

#define HD  128      // hidden dim
#define LDA 130      // LDS row stride in halfs (65-dword lane stride: bank-conflict free)

// K-offset of VGPR-pair j for the 16-bit A/B WMMA fragment layout (ISA 7.12.2):
// lanes 0-15: VGPR0..3 -> K=0..7, VGPR4..7 -> K=16..23 ; lanes 16-31: +8.
__device__ __forceinline__ int kbase_of(int j, int lane) {
    return (j < 4 ? 2 * j : 2 * j + 8) + ((lane & 16) ? 8 : 0);
}

// Stage a BM x 128 block of f32 activations into LDS as f16 (rows >= N -> 0).
__device__ __forceinline__ void stageA(const float* __restrict__ A, int row0, int N,
                                       _Float16* As, int BM, int tid, int tcount) {
    for (int idx = tid; idx < BM * HD; idx += tcount) {
        int r = idx >> 7, c = idx & 127;
        int gr = row0 + r;
        float x = (gr < N) ? A[(size_t)gr * HD + c] : 0.0f;
        As[r * LDA + c] = (_Float16)x;
    }
}

__device__ __forceinline__ v16h load_afrag(const _Float16* As, int rowbase, int lane, int k0) {
    v16h a;
    int r = rowbase + (lane & 15);
#pragma unroll
    for (int j = 0; j < 8; ++j) {
        int k = k0 + kbase_of(j, lane);
        a[2 * j]     = As[r * LDA + k];
        a[2 * j + 1] = As[r * LDA + k + 1];
    }
    return a;
}

__device__ __forceinline__ float sigmoidf_(float x) { return 1.0f / (1.0f + __expf(-x)); }

// ---------------------------------------------------------------------------
// Weight conversion: f32 row-major [K][128] -> fragment-swizzled f16 so that
// each B fragment is one contiguous 32-byte chunk per lane.
// ---------------------------------------------------------------------------
__global__ void k_wconv(const float* __restrict__ W, _Float16* __restrict__ Wf, int K) {
    int idx = blockIdx.x * blockDim.x + threadIdx.x;
    int total = (K / 32) << 11;              // (K/32)*8*32*8
    if (idx >= total) return;
    int j    = idx & 7;
    int lane = (idx >> 3) & 31;
    int ct   = (idx >> 8) & 7;
    int ks   = idx >> 11;
    int k    = ks * 32 + kbase_of(j, lane);
    int col  = ct * 16 + (lane & 15);
    size_t base = ((size_t)(ks * 8 + ct) * 32 + lane) * 16;
    Wf[base + 2 * j]     = (_Float16)W[(size_t)k * HD + col];
    Wf[base + 2 * j + 1] = (_Float16)W[(size_t)(k + 1) * HD + col];
}

// ---------------------------------------------------------------------------
// Generic GEMM: O = act(A[Nx128] @ W[128x128] + bias).  ACT: 0=none, 1=tanh.
// 256 threads = 8 waves; wave w computes rows [16w,16w+16) x all 128 cols.
// ---------------------------------------------------------------------------
template <int ACT>
__global__ __launch_bounds__(256)
void k_gemm128(const float* __restrict__ A, const _Float16* __restrict__ Wf,
               const float* __restrict__ bias, float* __restrict__ O, int N) {
    __shared__ _Float16 As[HD * LDA];
    int row0 = blockIdx.x * 128;
    stageA(A, row0, N, As, 128, threadIdx.x, 256);
    __syncthreads();

    int lane = threadIdx.x & 31;
    int w    = threadIdx.x >> 5;
    int wrow = w * 16;
    const v16h* Wfv = (const v16h*)Wf;

    v8f acc[8];
#pragma unroll
    for (int c = 0; c < 8; ++c) acc[c] = (v8f){0, 0, 0, 0, 0, 0, 0, 0};

#pragma unroll
    for (int ks = 0; ks < 4; ++ks) {
        v16h a = load_afrag(As, wrow, lane, ks * 32);
#pragma unroll
        for (int c = 0; c < 8; ++c) {
            v16h b = Wfv[(ks * 8 + c) * 32 + lane];
            acc[c] = __builtin_amdgcn_wmma_f32_16x16x32_f16(
                false, a, false, b, (short)0, acc[c], false, false);
        }
    }

    int rbase = wrow + ((lane & 16) ? 8 : 0);
#pragma unroll
    for (int c = 0; c < 8; ++c) {
        int col = c * 16 + (lane & 15);
        float bb = bias[col];
#pragma unroll
        for (int v = 0; v < 8; ++v) {
            int gr = row0 + rbase + v;
            if (gr < N) {
                float y = acc[c][v] + bb;
                if (ACT == 1) y = tanhf(y);
                O[(size_t)gr * HD + col] = y;
            }
        }
    }
}

// ---------------------------------------------------------------------------
// m += has_pred[r] * (inv_deg[r] * (aggr @ Wn) + bn)   (per-row scale commutes)
// ---------------------------------------------------------------------------
__global__ __launch_bounds__(256)
void k_gemm_na(const float* __restrict__ A, const _Float16* __restrict__ Wf,
               const float* __restrict__ bias, const float* __restrict__ inv_deg,
               const float* __restrict__ has_pred, float* __restrict__ O, int N) {
    __shared__ _Float16 As[HD * LDA];
    int row0 = blockIdx.x * 128;
    stageA(A, row0, N, As, 128, threadIdx.x, 256);
    __syncthreads();

    int lane = threadIdx.x & 31;
    int w    = threadIdx.x >> 5;
    int wrow = w * 16;
    const v16h* Wfv = (const v16h*)Wf;

    v8f acc[8];
#pragma unroll
    for (int c = 0; c < 8; ++c) acc[c] = (v8f){0, 0, 0, 0, 0, 0, 0, 0};

#pragma unroll
    for (int ks = 0; ks < 4; ++ks) {
        v16h a = load_afrag(As, wrow, lane, ks * 32);
#pragma unroll
        for (int c = 0; c < 8; ++c) {
            v16h b = Wfv[(ks * 8 + c) * 32 + lane];
            acc[c] = __builtin_amdgcn_wmma_f32_16x16x32_f16(
                false, a, false, b, (short)0, acc[c], false, false);
        }
    }

    int rbase = wrow + ((lane & 16) ? 8 : 0);
#pragma unroll
    for (int c = 0; c < 8; ++c) {
        int col = c * 16 + (lane & 15);
        float bb = bias[col];
#pragma unroll
        for (int v = 0; v < 8; ++v) {
            int gr = row0 + rbase + v;
            if (gr < N) {
                float g  = has_pred[gr];
                float iv = inv_deg[gr];
                size_t o = (size_t)gr * HD + col;
                O[o] = O[o] + g * (iv * acc[c][v] + bb);
            }
        }
    }
}

// ---------------------------------------------------------------------------
// Gate + LayerNorm + ReLU, fused.  K=256 concat GEMM [h, m] @ Wg + bg.
// BM=64, 128 threads = 4 waves; each wave owns 16 full rows, so LayerNorm
// reduces with intra-half-wave shuffles only (masks 1/2/4/8).
// ---------------------------------------------------------------------------
__global__ __launch_bounds__(128)
void k_gate_ln(const float* __restrict__ hin, const float* __restrict__ mbuf,
               const _Float16* __restrict__ Wf, const float* __restrict__ bg,
               const float* __restrict__ gamma, const float* __restrict__ beta,
               float* __restrict__ hout, int N) {
    __shared__ _Float16 hS[64 * LDA];
    __shared__ _Float16 mS[64 * LDA];
    int row0 = blockIdx.x * 64;
    stageA(hin,  row0, N, hS, 64, threadIdx.x, 128);
    stageA(mbuf, row0, N, mS, 64, threadIdx.x, 128);
    __syncthreads();

    int lane = threadIdx.x & 31;
    int w    = threadIdx.x >> 5;      // 0..3
    int wrow = w * 16;
    const v16h* Wfv = (const v16h*)Wf;

    v8f acc[8];
#pragma unroll
    for (int c = 0; c < 8; ++c) acc[c] = (v8f){0, 0, 0, 0, 0, 0, 0, 0};

#pragma unroll
    for (int ks = 0; ks < 8; ++ks) {               // K = 256
        const _Float16* As = (ks < 4) ? hS : mS;
        v16h a = load_afrag(As, wrow, lane, (ks & 3) * 32);
#pragma unroll
        for (int c = 0; c < 8; ++c) {
            v16h b = Wfv[(ks * 8 + c) * 32 + lane];
            acc[c] = __builtin_amdgcn_wmma_f32_16x16x32_f16(
                false, a, false, b, (short)0, acc[c], false, false);
        }
    }

    int rbase = wrow + ((lane & 16) ? 8 : 0);
    // v = sigmoid(lin)*m + (1-sigmoid)*h ; overwrite acc with v
#pragma unroll
    for (int c = 0; c < 8; ++c) {
        int col = c * 16 + (lane & 15);
        float bb = bg[col];
#pragma unroll
        for (int v = 0; v < 8; ++v) {
            int rl = rbase + v;
            float g  = sigmoidf_(acc[c][v] + bb);
            float hv = (float)hS[rl * LDA + col];
            float mv = (float)mS[rl * LDA + col];
            acc[c][v] = g * mv + (1.0f - g) * hv;
        }
    }
    // LayerNorm + ReLU per row (row lives entirely in one 16-lane half)
#pragma unroll
    for (int v = 0; v < 8; ++v) {
        float s = 0.f, s2 = 0.f;
#pragma unroll
        for (int c = 0; c < 8; ++c) { float x = acc[c][v]; s += x; s2 += x * x; }
#pragma unroll
        for (int mk = 8; mk >= 1; mk >>= 1) {
            s  += __shfl_xor(s,  mk, 32);
            s2 += __shfl_xor(s2, mk, 32);
        }
        float mu  = s * (1.0f / 128.0f);
        float var = s2 * (1.0f / 128.0f) - mu * mu;
        float rs  = rsqrtf(var + 1e-5f);
        int gr = row0 + rbase + v;
        if (gr < N) {
#pragma unroll
            for (int c = 0; c < 8; ++c) {
                int col = c * 16 + (lane & 15);
                float y = (acc[c][v] - mu) * rs * gamma[col] + beta[col];
                hout[(size_t)gr * HD + col] = fmaxf(y, 0.0f);
            }
        }
    }
}

// ---------------------------------------------------------------------------
// Edge aggregation + degree bookkeeping
// ---------------------------------------------------------------------------
__global__ void k_zero(float* p, size_t n) {
    size_t i = (size_t)blockIdx.x * blockDim.x + threadIdx.x;
    size_t st = (size_t)gridDim.x * blockDim.x;
    for (; i < n; i += st) p[i] = 0.0f;
}

__global__ void k_deg(const int* __restrict__ src, const int* __restrict__ dst,
                      float* indeg, float* outdeg, int E) {
    int i = blockIdx.x * blockDim.x + threadIdx.x;
    if (i < E) {
        atomicAdd(&indeg[dst[i]], 1.0f);
        atomicAdd(&outdeg[src[i]], 1.0f);
    }
}

__global__ void k_degfin(const float* __restrict__ indeg, const float* __restrict__ outdeg,
                         float* inv_deg, float* has_pred, float* sink, int N) {
    int i = blockIdx.x * blockDim.x + threadIdx.x;
    if (i < N) {
        float d = indeg[i];
        inv_deg[i]  = 1.0f / fmaxf(d, 1.0f);
        has_pred[i] = (d > 0.0f) ? 1.0f : 0.0f;
        sink[i]     = (outdeg[i] == 0.0f) ? 1.0f : 0.0f;
    }
}

// One wave per edge: float4 gather of h[src], 4 f32 atomics into aggr[dst].
__global__ __launch_bounds__(256)
void k_scatter(const float* __restrict__ h, const int* __restrict__ src,
               const int* __restrict__ dst, float* __restrict__ aggr, int E) {
    int gtid = blockIdx.x * blockDim.x + threadIdx.x;
    int e    = gtid >> 5;
    int lane = gtid & 31;
    if (e >= E) return;
    int s = src[e], d = dst[e];
    float4 x = ((const float4*)(h + (size_t)s * HD))[lane];
    float* o = aggr + (size_t)d * HD + lane * 4;
    atomicAdd(o + 0, x.x);
    atomicAdd(o + 1, x.y);
    atomicAdd(o + 2, x.z);
    atomicAdd(o + 3, x.w);
}

// ---------------------------------------------------------------------------
// Attention pooling. tanh-bounded scores -> skip max pass (exp is safe,
// non-sinks get weight 0, identical to reference softmax mathematically).
// ---------------------------------------------------------------------------
__global__ __launch_bounds__(256)
void k_score(const float* __restrict__ a, const float* __restrict__ Wsc,
             const float* __restrict__ bsc, const float* __restrict__ sink,
             float* __restrict__ p, float* __restrict__ Z, int N) {
    int gtid = blockIdx.x * blockDim.x + threadIdx.x;
    int n    = gtid >> 5;
    int lane = gtid & 31;
    if (n >= N) return;
    float4 x = ((const float4*)(a + (size_t)n * HD))[lane];
    float4 w = ((const float4*)Wsc)[lane];
    float s = x.x * w.x + x.y * w.y + x.z * w.z + x.w * w.w;
#pragma unroll
    for (int mk = 16; mk >= 1; mk >>= 1) s += __shfl_xor(s, mk, 32);
    if (lane == 0) {
        float sc = s + bsc[0];
        float pv = (sink[n] > 0.5f) ? __expf(sc) : 0.0f;
        p[n] = pv;
        if (pv > 0.0f) atomicAdd(Z, pv);
    }
}

__global__ __launch_bounds__(256)
void k_emb(const float* __restrict__ h, const float* __restrict__ p,
           const float* __restrict__ Z, float* __restrict__ emb, int N) {
    __shared__ float accs[HD];
    if (threadIdx.x < HD) accs[threadIdx.x] = 0.0f;
    __syncthreads();
    int col  = threadIdx.x & 127;
    int rofs = threadIdx.x >> 7;
    int n0   = blockIdx.x * 256;
    float local = 0.0f;
    for (int r = rofs; r < 256; r += 2) {
        int n = n0 + r;
        if (n < N) {
            float pv = p[n];
            if (pv != 0.0f) local += pv * h[(size_t)n * HD + col];
        }
    }
    if (local != 0.0f) atomicAdd(&accs[col], local);
    __syncthreads();
    if (threadIdx.x < HD) {
        float v = accs[threadIdx.x];
        if (v != 0.0f) atomicAdd(&emb[threadIdx.x], v / Z[0]);
    }
}

// ---------------------------------------------------------------------------
// Host orchestration
// ---------------------------------------------------------------------------
extern "C" void kernel_launch(void* const* d_in, const int* in_sizes, int n_in,
                              void* d_out, int out_size, void* d_ws, size_t ws_size,
                              hipStream_t stream) {
    const float* node_feats = (const float*)d_in[0];
    const int*   src        = (const int*)d_in[1];
    const int*   dst        = (const int*)d_in[2];
    const float* W_in       = (const float*)d_in[3];
    const float* b_in       = (const float*)d_in[4];
    const float* Ws         = (const float*)d_in[5];
    const float* bs         = (const float*)d_in[6];
    const float* Wn         = (const float*)d_in[7];
    const float* bn         = (const float*)d_in[8];
    const float* Wg         = (const float*)d_in[9];
    const float* bg         = (const float*)d_in[10];
    const float* gamma      = (const float*)d_in[11];
    const float* beta       = (const float*)d_in[12];
    const float* W_att      = (const float*)d_in[13];
    const float* b_att      = (const float*)d_in[14];
    const float* W_score    = (const float*)d_in[15];
    const float* b_score    = (const float*)d_in[16];

    const int N = in_sizes[0] / HD;
    const int E = in_sizes[1];
    const int L = in_sizes[6] / HD;    // bs is (L, H)

    float* hbuf = (float*)d_out;                 // running h, updated in place
    float* emb  = hbuf + (size_t)N * HD;         // graph embedding tail

    // --- carve workspace ---
    char* cur = (char*)d_ws;
    auto carve = [&](size_t bytes) -> void* {
        void* pp = (void*)cur;
        cur += (bytes + 255) & ~(size_t)255;
        return pp;
    };
    float*     aggr  = (float*)carve((size_t)N * HD * 4);
    float*     mbuf  = (float*)carve((size_t)N * HD * 4);
    float*     deg2  = (float*)carve((size_t)2 * N * 4);  // indeg | outdeg
    float*     indeg = deg2, *outdeg = deg2 + N;
    float*     inv_d = (float*)carve((size_t)N * 4);
    float*     hpred = (float*)carve((size_t)N * 4);
    float*     sink  = (float*)carve((size_t)N * 4);
    float*     pbuf  = (float*)carve((size_t)N * 4);
    float*     Zbuf  = (float*)carve(256);
    _Float16*  WfIn  = (_Float16*)carve((size_t)HD * HD * 2);
    _Float16*  WfS   = (_Float16*)carve((size_t)L * HD * HD * 2);
    _Float16*  WfN   = (_Float16*)carve((size_t)L * HD * HD * 2);
    _Float16*  WfG   = (_Float16*)carve((size_t)L * 2 * HD * HD * 2);
    _Float16*  WfAtt = (_Float16*)carve((size_t)HD * HD * 2);

    const int NB128 = (N + 127) / 128;
    const int NB64  = (N + 63) / 64;

    // --- init per-call state ---
    k_zero<<<512, 256, 0, stream>>>(deg2, (size_t)2 * N);
    k_zero<<<1, 256, 0, stream>>>(Zbuf, 64);
    k_zero<<<1, 128, 0, stream>>>(emb, HD);

    // --- weight fragment conversion (f32 -> swizzled f16) ---
    k_wconv<<<32, 256, 0, stream>>>(W_in, WfIn, HD);
    for (int l = 0; l < L; ++l) {
        k_wconv<<<32, 256, 0, stream>>>(Ws + (size_t)l * HD * HD, WfS + (size_t)l * HD * HD, HD);
        k_wconv<<<32, 256, 0, stream>>>(Wn + (size_t)l * HD * HD, WfN + (size_t)l * HD * HD, HD);
        k_wconv<<<64, 256, 0, stream>>>(Wg + (size_t)l * 2 * HD * HD, WfG + (size_t)l * 2 * HD * HD, 2 * HD);
    }
    k_wconv<<<32, 256, 0, stream>>>(W_att, WfAtt, HD);

    // --- degrees ---
    k_deg<<<(E + 255) / 256, 256, 0, stream>>>(src, dst, indeg, outdeg, E);
    k_degfin<<<(N + 255) / 256, 256, 0, stream>>>(indeg, outdeg, inv_d, hpred, sink, N);

    // --- input projection: h = X @ W_in + b_in ---
    k_gemm128<0><<<NB128, 256, 0, stream>>>(node_feats, WfIn, b_in, hbuf, N);

    // --- layers ---
    for (int l = 0; l < L; ++l) {
        k_zero<<<2048, 256, 0, stream>>>(aggr, (size_t)N * HD);
        k_scatter<<<(E * 32 + 255) / 256, 256, 0, stream>>>(hbuf, src, dst, aggr, E);
        // m = h @ Ws + bs
        k_gemm128<0><<<NB128, 256, 0, stream>>>(hbuf, WfS + (size_t)l * HD * HD,
                                                bs + (size_t)l * HD, mbuf, N);
        // m += has_pred * (inv_deg * (aggr @ Wn) + bn)
        k_gemm_na<<<NB128, 256, 0, stream>>>(aggr, WfN + (size_t)l * HD * HD,
                                             bn + (size_t)l * HD, inv_d, hpred, mbuf, N);
        // gate + mix + LayerNorm + ReLU -> h (in place)
        k_gate_ln<<<NB64, 128, 0, stream>>>(hbuf, mbuf, WfG + (size_t)l * 2 * HD * HD,
                                            bg + (size_t)l * HD, gamma + (size_t)l * HD,
                                            beta + (size_t)l * HD, hbuf, N);
    }

    // --- attention pooling ---
    k_gemm128<1><<<NB128, 256, 0, stream>>>(hbuf, WfAtt, b_att, mbuf, N);  // a = tanh(...)
    k_score<<<(N * 32 + 255) / 256, 256, 0, stream>>>(mbuf, W_score, b_score, sink,
                                                      pbuf, Zbuf, N);
    k_emb<<<(N + 255) / 256, 256, 0, stream>>>(hbuf, pbuf, Zbuf, emb, N);
}
